// TemporalAttentionLayer_41102837023001
// MI455X (gfx1250) — matrix operand order repaired and checked by reading, not compiled, not run on hardware
//
#include <hip/hip_runtime.h>
#include <stdint.h>

#define D_DIM 1024
#define N_DIM 4096
#define B_DIM 4
#define BN    (B_DIM * N_DIM)
#define SCALE 0.03125f   /* 1/sqrt(1024) */
#define LN_EPS 1e-5f

typedef __bf16 bf16;
typedef bf16  v16bf __attribute__((ext_vector_type(16)));
typedef bf16  v8bf  __attribute__((ext_vector_type(8)));
typedef float v8f   __attribute__((ext_vector_type(8)));
typedef unsigned short v16us __attribute__((ext_vector_type(16)));
typedef int   v4i   __attribute__((ext_vector_type(4)));
typedef v4i __attribute__((address_space(1))) v4i_g;   // global
typedef v4i __attribute__((address_space(3))) v4i_l;   // LDS

#if defined(__has_builtin)
#if __has_builtin(__builtin_amdgcn_global_load_async_to_lds_b128)
#define HAVE_ASYNC_LDS 1
#endif
#endif

__device__ __forceinline__ unsigned short f2bf(float f) {
  unsigned int u = __float_as_uint(f);
  u += 0x7FFFu + ((u >> 16) & 1u);      // round-to-nearest-even
  return (unsigned short)(u >> 16);
}

__device__ __forceinline__ v8f vzero() {
  v8f v;
#pragma unroll
  for (int i = 0; i < 8; ++i) v[i] = 0.f;
  return v;
}

// bf16 A-fragment (16x32) in this file's contiguous k order (B matches): lane L<16 holds row L, K = {hf*16..hf*16+15}
__device__ __forceinline__ v16bf load_a_frag(const unsigned short* rowp, int hf) {
  v8bf lo = *(const v8bf*)(rowp + hf * 16);
  v8bf hi = *(const v8bf*)(rowp + hf * 16 + 8);
  v16bf a;
#pragma unroll
  for (int i = 0; i < 8; ++i) { a[i] = lo[i]; a[i + 8] = hi[i]; }
  return a;
}

__device__ __forceinline__ v8f wmma_bf16(v16bf a, v16bf b, v8f c) {
  return __builtin_amdgcn_wmma_f32_16x16x32_bf16(false, a, false, b, (short)0, c,
                                                 false, false);
}

// ---------------------------------------------------------------------------
// Kernel 1: Q/K/V projection GEMM, y = x W^T + b, bf16 outputs.
// Block tile 128(M) x 128(N), K-step 64, 8 waves (2x4), each wave 64x32.
// z = blockIdx.z selects Q/K/V. V is written transposed [b][d][n].
// ---------------------------------------------------------------------------
__global__ __launch_bounds__(256) void qkv_gemm_kernel(
    const float* __restrict__ X,
    const float* __restrict__ Wq, const float* __restrict__ Wk,
    const float* __restrict__ Wv,
    const float* __restrict__ bq, const float* __restrict__ bk,
    const float* __restrict__ bv,
    unsigned short* __restrict__ Qb, unsigned short* __restrict__ Kb,
    unsigned short* __restrict__ Vt)
{
  __shared__ unsigned short Xlds[128][64];
  __shared__ unsigned short Wlds[128][64];
  const int tid  = threadIdx.x;
  const int w    = tid >> 5, lane = tid & 31, l16 = lane & 15, hf = lane >> 4;
  const int wm   = w & 1, wn = w >> 1;
  const int gm0  = blockIdx.x * 128;
  const int gn0  = blockIdx.y * 128;
  const int z    = blockIdx.z;
  const float* Wsel = (z == 0) ? Wq : (z == 1) ? Wk : Wv;
  const float* bsel = (z == 0) ? bq : (z == 1) ? bk : bv;

  v8f acc[4][2];
#pragma unroll
  for (int i = 0; i < 4; ++i)
#pragma unroll
    for (int j = 0; j < 2; ++j) acc[i][j] = vzero();

  for (int kk = 0; kk < D_DIM; kk += 64) {
    __syncthreads();
#pragma unroll
    for (int j = 0; j < 8; ++j) {
      int p = tid + j * 256;
      int row = p >> 4, c4 = p & 15;
      float4 xv = *(const float4*)(X + (size_t)(gm0 + row) * D_DIM + kk + c4 * 4);
      float4 wv = *(const float4*)(Wsel + (size_t)(gn0 + row) * D_DIM + kk + c4 * 4);
      uint2 xp, wp;
      xp.x = f2bf(xv.x) | ((unsigned)f2bf(xv.y) << 16);
      xp.y = f2bf(xv.z) | ((unsigned)f2bf(xv.w) << 16);
      wp.x = f2bf(wv.x) | ((unsigned)f2bf(wv.y) << 16);
      wp.y = f2bf(wv.z) | ((unsigned)f2bf(wv.w) << 16);
      *(uint2*)&Xlds[row][c4 * 4] = xp;
      *(uint2*)&Wlds[row][c4 * 4] = wp;
    }
    __syncthreads();

#pragma unroll
    for (int ks = 0; ks < 64; ks += 32) {
      v16bf a[4], b[2];
#pragma unroll
      for (int i = 0; i < 4; ++i)
        a[i] = load_a_frag(&Xlds[wm * 64 + i * 16 + l16][ks], hf);
#pragma unroll
      for (int j = 0; j < 2; ++j)
        b[j] = *(const v16bf*)&Wlds[wn * 32 + j * 16 + l16][ks + hf * 16];
#pragma unroll
      for (int i = 0; i < 4; ++i)
#pragma unroll
        for (int j = 0; j < 2; ++j)
          acc[i][j] = wmma_bf16(a[i], b[j], acc[i][j]);
    }
  }

  // epilogue: + bias, cast bf16, store (C layout: VGPR r -> row r+8*hf, col = l16)
#pragma unroll
  for (int i = 0; i < 4; ++i) {
#pragma unroll
    for (int j = 0; j < 2; ++j) {
      int gcol = gn0 + wn * 32 + j * 16 + l16;
      float bias = bsel[gcol];
#pragma unroll
      for (int r = 0; r < 8; ++r) {
        int grow = gm0 + wm * 64 + i * 16 + r + 8 * hf;
        unsigned short h = f2bf(acc[i][j][r] + bias);
        if (z == 2) {
          int bb = grow >> 12, nn = grow & (N_DIM - 1);
          Vt[((size_t)bb * D_DIM + gcol) * N_DIM + nn] = h;
        } else {
          unsigned short* O = (z == 0) ? Qb : Kb;
          O[(size_t)grow * D_DIM + gcol] = h;
        }
      }
    }
  }
}

// ---------------------------------------------------------------------------
// Kernel 2: softmax statistics (pass 1). 4 waves x 16 rows per block.
// K chunks (32 keys x 1024) staged in 64KB LDS via async global->LDS copies
// (ASYNCcnt path) when the toolchain exposes the builtin.
// ---------------------------------------------------------------------------
__global__ __launch_bounds__(128) void attn_stats_kernel(
    const unsigned short* __restrict__ Qb, const unsigned short* __restrict__ Kb,
    float* __restrict__ Mrow, float* __restrict__ Srow)
{
  __shared__ unsigned short Klds[32][1024];
  const int tid  = threadIdx.x;
  const int w    = tid >> 5, lane = tid & 31, l16 = lane & 15, hf = lane >> 4;
  const int n0   = blockIdx.x * 64 + w * 16;
  const int bidx = n0 >> 12;
  const size_t kb = ((size_t)bidx << 12) * D_DIM;
  const unsigned short* qrow = Qb + (size_t)(n0 + l16) * D_DIM;

  float mx[8], sm[8];
#pragma unroll
  for (int r = 0; r < 8; ++r) { mx[r] = -1e30f; sm[r] = 0.f; }

  for (int m0 = 0; m0 < N_DIM; m0 += 32) {
    __syncthreads();
    const uint4* src = (const uint4*)(Kb + kb + (size_t)m0 * D_DIM);
#ifdef HAVE_ASYNC_LDS
#pragma unroll
    for (int j = 0; j < 16; ++j) {
      int p = tid + j * 128;
      int row = p >> 6, c = p & 63;
      __builtin_amdgcn_global_load_async_to_lds_b128(
          (v4i_g*)(src + row * 128 + c),
          (v4i_l*)&Klds[row][c * 8], 0, 0);
    }
#if __has_builtin(__builtin_amdgcn_s_wait_asynccnt)
    __builtin_amdgcn_s_wait_asynccnt(0);
#else
    asm volatile("s_wait_asynccnt 0x0" ::: "memory");
#endif
#else
#pragma unroll
    for (int j = 0; j < 16; ++j) {
      int p = tid + j * 128;
      int row = p >> 6, c = p & 63;
      *(uint4*)&Klds[row][c * 8] = src[row * 128 + c];
    }
#endif
    if (m0 + 32 < N_DIM)  // prefetch next K chunk (global_prefetch_b8)
      __builtin_prefetch((const char*)(src + 4096) + tid * 512, 0, 1);
    __syncthreads();

    v8f c0 = vzero(), c1 = vzero();
#pragma unroll 4
    for (int kk = 0; kk < D_DIM; kk += 32) {
      v16bf a  = load_a_frag(qrow + kk, hf);
      v16bf b0 = *(const v16bf*)&Klds[l16][kk + hf * 16];
      v16bf b1 = *(const v16bf*)&Klds[16 + l16][kk + hf * 16];
      c0 = wmma_bf16(a, b0, c0);
      c1 = wmma_bf16(a, b1, c1);
    }
    // online softmax stats, per C-row r (row = n0 + r + 8*hf)
#pragma unroll
    for (int r = 0; r < 8; ++r) {
      float v0 = c0[r] * SCALE, v1 = c1[r] * SCALE;
      float vm = fmaxf(v0, v1);
#pragma unroll
      for (int msk = 8; msk >= 1; msk >>= 1)
        vm = fmaxf(vm, __shfl_xor(vm, msk, 32));
      float nm = fmaxf(mx[r], vm);
      float es = __expf(v0 - nm) + __expf(v1 - nm);
#pragma unroll
      for (int msk = 8; msk >= 1; msk >>= 1)
        es += __shfl_xor(es, msk, 32);
      sm[r] = sm[r] * __expf(mx[r] - nm) + es;
      mx[r] = nm;
    }
  }
  if (l16 == 0) {
#pragma unroll
    for (int r = 0; r < 8; ++r) {
      Mrow[n0 + 8 * hf + r] = mx[r];
      Srow[n0 + 8 * hf + r] = sm[r];
    }
  }
}

// ---------------------------------------------------------------------------
// Kernel 3: attention output + residual + LayerNorm (pass 2).
// One 16-row tile per block; 4 waves split D (256 each). Partial score tiles
// combined across waves with ds_add_f32 atomics in LDS.
// ---------------------------------------------------------------------------
__global__ __launch_bounds__(128) void attn_out_kernel(
    const float* __restrict__ emb,
    const unsigned short* __restrict__ Qb, const unsigned short* __restrict__ Kb,
    const unsigned short* __restrict__ Vt,
    const float* __restrict__ Mrow, const float* __restrict__ Srow,
    const float* __restrict__ gvec, const float* __restrict__ bvec,
    float* __restrict__ outp)
{
  __shared__ float Stile[16 * 32];
  __shared__ float RowAcc[32];
  const int tid  = threadIdx.x;
  const int w    = tid >> 5, lane = tid & 31, l16 = lane & 15, hf = lane >> 4;
  const int n0   = blockIdx.x * 16;
  const int bidx = n0 >> 12;
  const int d0   = w * 256;
  const size_t kb  = ((size_t)bidx << 12) * D_DIM;
  const size_t vtb = (size_t)bidx * D_DIM * N_DIM;
  if (tid < 32) RowAcc[tid] = 0.f;

  v8f o[16];
#pragma unroll
  for (int t = 0; t < 16; ++t) o[t] = vzero();

  const float Mr = Mrow[n0 + l16];
  const unsigned short* qrow = Qb + (size_t)(n0 + l16) * D_DIM + d0;

  for (int m0 = 0; m0 < N_DIM; m0 += 32) {
    for (int p = tid; p < 512; p += 128) Stile[p] = 0.f;
    __syncthreads();

    // partial scores over this wave's d-range [d0, d0+256)
    v8f c0 = vzero(), c1 = vzero();
    const unsigned short* k0p = Kb + kb + (size_t)(m0 + l16) * D_DIM + d0;
    const unsigned short* k1p = k0p + (size_t)16 * D_DIM;
#pragma unroll
    for (int kk = 0; kk < 256; kk += 32) {
      v16bf a  = load_a_frag(qrow + kk, hf);
      v16bf b0 = *(const v16bf*)(k0p + kk + hf * 16);
      v16bf b1 = *(const v16bf*)(k1p + kk + hf * 16);
      c0 = wmma_bf16(a, b0, c0);
      c1 = wmma_bf16(a, b1, c1);
    }
#pragma unroll
    for (int r = 0; r < 8; ++r) {
      atomicAdd(&Stile[(r + 8 * hf) * 32 + l16],      c0[r] * SCALE);
      atomicAdd(&Stile[(r + 8 * hf) * 32 + 16 + l16], c1[r] * SCALE);
    }
    __syncthreads();

    // rebuild P = exp(s - M) as a bf16 A-fragment (row = l16)
    v16us pu;
#pragma unroll
    for (int i = 0; i < 8; ++i) {
      float s0 = Stile[l16 * 32 + hf * 16 + i];
      float s1 = Stile[l16 * 32 + hf * 16 + 8 + i];
      pu[i]     = f2bf(__expf(s0 - Mr));
      pu[i + 8] = f2bf(__expf(s1 - Mr));
    }
    v16bf a2 = __builtin_bit_cast(v16bf, pu);

    // O += P @ V over this wave's 16 d-tiles; Vt rows are m-contiguous
#pragma unroll
    for (int t = 0; t < 16; ++t) {
      const unsigned short* vp =
          Vt + vtb + (size_t)(d0 + t * 16 + l16) * N_DIM + m0 + hf * 16;
      v16bf bv = *(const v16bf*)vp;
      o[t] = wmma_bf16(a2, bv, o[t]);
    }
    __syncthreads();
  }

  // epilogue: x = emb + O/S, then LayerNorm over full D via LDS reduction
  float sr[8], xs[8], xq[8];
#pragma unroll
  for (int r = 0; r < 8; ++r) {
    sr[r] = 1.f / Srow[n0 + r + 8 * hf];
    xs[r] = 0.f; xq[r] = 0.f;
  }
#pragma unroll
  for (int t = 0; t < 16; ++t) {
#pragma unroll
    for (int r = 0; r < 8; ++r) {
      int row = n0 + r + 8 * hf;
      int col = d0 + t * 16 + l16;
      float x = emb[(size_t)row * D_DIM + col] + o[t][r] * sr[r];
      o[t][r] = x;
      xs[r] += x; xq[r] += x * x;
    }
  }
#pragma unroll
  for (int r = 0; r < 8; ++r) {
#pragma unroll
    for (int msk = 8; msk >= 1; msk >>= 1) {
      xs[r] += __shfl_xor(xs[r], msk, 32);
      xq[r] += __shfl_xor(xq[r], msk, 32);
    }
  }
  if (l16 == 0) {
#pragma unroll
    for (int r = 0; r < 8; ++r) {
      atomicAdd(&RowAcc[(r + 8 * hf) * 2],     xs[r]);
      atomicAdd(&RowAcc[(r + 8 * hf) * 2 + 1], xq[r]);
    }
  }
  __syncthreads();

  float mu[8], rs[8];
#pragma unroll
  for (int r = 0; r < 8; ++r) {
    float s = RowAcc[(r + 8 * hf) * 2];
    float q = RowAcc[(r + 8 * hf) * 2 + 1];
    float m = s * (1.f / D_DIM);
    mu[r] = m;
    rs[r] = rsqrtf(q * (1.f / D_DIM) - m * m + LN_EPS);
  }
#pragma unroll
  for (int t = 0; t < 16; ++t) {
    int col = d0 + t * 16 + l16;
    float g = gvec[col], be = bvec[col];
#pragma unroll
    for (int r = 0; r < 8; ++r) {
      int row = n0 + r + 8 * hf;
      outp[(size_t)row * D_DIM + col] = g * (o[t][r] - mu[r]) * rs[r] + be;
    }
  }
}

// ---------------------------------------------------------------------------
extern "C" void kernel_launch(void* const* d_in, const int* in_sizes, int n_in,
                              void* d_out, int out_size, void* d_ws, size_t ws_size,
                              hipStream_t stream) {
  (void)in_sizes; (void)n_in; (void)out_size; (void)ws_size;
  const float* emb  = (const float*)d_in[0];
  // d_in[1..3] (edge_indices, edge_times, timestamps) unused by the reference.
  const float* Wq   = (const float*)d_in[4];
  const float* bq   = (const float*)d_in[5];
  const float* Wk   = (const float*)d_in[6];
  const float* bk   = (const float*)d_in[7];
  const float* Wv   = (const float*)d_in[8];
  const float* bv   = (const float*)d_in[9];
  const float* gvec = (const float*)d_in[10];
  const float* bvec = (const float*)d_in[11];
  float* outp = (float*)d_out;

  // workspace: Q, K (bf16 [b*n][d]), V transposed (bf16 [b][d][n]), stats
  unsigned short* Qb = (unsigned short*)d_ws;
  unsigned short* Kb = Qb + (size_t)BN * D_DIM;
  unsigned short* Vt = Kb + (size_t)BN * D_DIM;
  float* Mrow = (float*)(Vt + (size_t)BN * D_DIM);
  float* Srow = Mrow + BN;

  qkv_gemm_kernel<<<dim3(BN / 128, D_DIM / 128, 3), 256, 0, stream>>>(
      emb, Wq, Wk, Wv, bq, bk, bv, Qb, Kb, Vt);
  attn_stats_kernel<<<dim3(BN / 64), 128, 0, stream>>>(Qb, Kb, Mrow, Srow);
  attn_out_kernel<<<dim3(BN / 16), 128, 0, stream>>>(
      emb, Qb, Kb, Vt, Mrow, Srow, gvec, bvec, outp);
}
